// MiniCPMOMultiModalResampler_53291954208981
// MI455X (gfx1250) — matrix operand order repaired and checked by reading, not compiled, not run on hardware
//
#include <hip/hip_runtime.h>

// ---------------------------------------------------------------------------
// MiniCPM-O multimodal resampler for MI455X (gfx1250, wave32, WMMA bf16)
//   - big GEMMs: TDM (tensor_load_to_lds) double-buffered B-tile staging in
//     LDS with pad-injected 80B row pitch (bank-conflict-free ds_load_b128),
//     A operands direct from global (compiler pipelines with s_clause).
//   - attention GEMMs: direct-global WMMA (K small, operands per (b,h)).
// All GEMM accumulation f32, operands bf16.
// ---------------------------------------------------------------------------

typedef __bf16 bf16;
typedef __bf16 v16bf __attribute__((ext_vector_type(16)));
typedef float  f32x8 __attribute__((ext_vector_type(8)));
typedef float  f32x4 __attribute__((ext_vector_type(4)));
typedef unsigned int u32x4 __attribute__((ext_vector_type(4)));
typedef int    i32x8 __attribute__((ext_vector_type(8)));
typedef int    i32x4 __attribute__((ext_vector_type(4)));

constexpr int  EDIM = 2048;
constexpr int  VPM  = 1152;
constexpr int  NQ   = 64;
constexpr int  NH   = 16;
constexpr int  LSEQ = 1024;   // 32*32
constexpr int  BATCH = 32;

// ---------------- workspace layout (bytes, lifetime-aliased) ---------------
constexpr size_t OFF_IMGBF = 0;                              // 75,497,472
constexpr size_t OFF_KVWBF = 75497472;                       //  4,718,592
constexpr size_t OFF_OBF   = 0;                              //  8,388,608 (after GEMM1)
constexpr size_t OFF_O2F   = 8388608;                        // 16,777,216
constexpr size_t OFF_O2LN  = 25165824;                       //  8,388,608
constexpr size_t OFF_W     = 80216064;                       // weights bf16
constexpr size_t OFF_WQ    = OFF_W;
constexpr size_t OFF_WK    = OFF_W + 1ull * 8388608;
constexpr size_t OFF_WV    = OFF_W + 2ull * 8388608;
constexpr size_t OFF_WO    = OFF_W + 3ull * 8388608;
constexpr size_t OFF_PJT   = OFF_W + 4ull * 8388608;
constexpr size_t OFF_QLN   = 122159104;                      // 262,144
constexpr size_t OFF_QH    = 122421248;                      // 262,144
constexpr size_t OFF_S1    = 122683392;  // xBF  -> khBF      134,217,728
constexpr size_t OFF_S2    = 256901120;  // kBF  -> scoresF   134,217,728
constexpr size_t OFF_S3    = 391118848;  // vBF  -> attnBF    134,217,728
constexpr size_t OFF_S4    = 525336576;  // vhT               134,217,728
// total ~659.6 MB

// ------------------------------- converts ----------------------------------
__global__ void k_cvt_bf16(const float* __restrict__ in, bf16* __restrict__ out, long n) {
  long i = (long)blockIdx.x * blockDim.x + threadIdx.x;
  long stride = (long)gridDim.x * blockDim.x;
  for (; i < n; i += stride) out[i] = (bf16)in[i];
}

__global__ void k_cvt_transpose(const float* __restrict__ in, bf16* __restrict__ out) {
  __shared__ float tile[32][33];
  int bx = blockIdx.x * 32, by = blockIdx.y * 32;
  int tx = threadIdx.x & 31, ty = threadIdx.x >> 5;   // 256 thr = 32x8
  for (int r = ty; r < 32; r += 8)
    tile[r][tx] = in[(long)(by + r) * EDIM + bx + tx];
  __syncthreads();
  for (int r = ty; r < 32; r += 8)
    out[(long)(bx + r) * EDIM + by + tx] = (bf16)tile[tx][r];
}

// --------------------------- block reductions ------------------------------
__device__ __forceinline__ float blk_sum(float v, float* sbuf) {
  int t = threadIdx.x;
  sbuf[t] = v; __syncthreads();
  for (int s = 128; s > 0; s >>= 1) { if (t < s) sbuf[t] += sbuf[t + s]; __syncthreads(); }
  float r = sbuf[0]; __syncthreads(); return r;
}
__device__ __forceinline__ float blk_max(float v, float* sbuf) {
  int t = threadIdx.x;
  sbuf[t] = v; __syncthreads();
  for (int s = 128; s > 0; s >>= 1) { if (t < s) sbuf[t] = fmaxf(sbuf[t], sbuf[t + s]); __syncthreads(); }
  float r = sbuf[0]; __syncthreads(); return r;
}

// ------------------------- layernorm (f32 in) ------------------------------
__global__ void k_ln_f32(const float* __restrict__ in, const float* __restrict__ g,
                         const float* __restrict__ bb, bf16* __restrict__ out) {
  __shared__ float sbuf[256];
  long row = blockIdx.x;
  const float* x = in + row * EDIM;
  float v[8];
  float s = 0.f;
#pragma unroll
  for (int t = 0; t < 8; t++) { v[t] = x[threadIdx.x + t * 256]; s += v[t]; }
  float mu = blk_sum(s, sbuf) * (1.0f / EDIM);
  float q = 0.f;
#pragma unroll
  for (int t = 0; t < 8; t++) { float d = v[t] - mu; q += d * d; }
  float rs = rsqrtf(blk_sum(q, sbuf) * (1.0f / EDIM) + 1e-5f);
#pragma unroll
  for (int t = 0; t < 8; t++) {
    int i = threadIdx.x + t * 256;
    out[row * EDIM + i] = (bf16)((v[t] - mu) * rs * g[i] + bb[i]);
  }
}

// -------------- ln_kv + 2D sincos positional embedding ---------------------
__global__ void k_ln_kv_pe(const bf16* __restrict__ xin, const float* __restrict__ g,
                           const float* __restrict__ bb, const int* __restrict__ tgt,
                           bf16* __restrict__ vout, bf16* __restrict__ kout) {
  __shared__ float sbuf[256];
  long row = blockIdx.x;                 // b*1024 + l
  int b = (int)(row >> 10), l = (int)(row & 1023);
  int th = tgt[2 * b];
  int tw = tgt[2 * b + 1]; tw = tw < 1 ? 1 : (tw > 32 ? 32 : tw);
  int rpos = l / tw; rpos = rpos > 31 ? 31 : rpos;
  int cpos = l % tw;
  bool valid = l < th * tw;

  float v[8];
  float s = 0.f;
#pragma unroll
  for (int t = 0; t < 8; t++) { v[t] = (float)xin[row * EDIM + threadIdx.x + t * 256]; s += v[t]; }
  float mu = blk_sum(s, sbuf) * (1.0f / EDIM);
  float q = 0.f;
#pragma unroll
  for (int t = 0; t < 8; t++) { float d = v[t] - mu; q += d * d; }
  float rs = rsqrtf(blk_sum(q, sbuf) * (1.0f / EDIM) + 1e-5f);
#pragma unroll
  for (int t = 0; t < 8; t++) {
    int e = threadIdx.x + t * 256;
    float y = (v[t] - mu) * rs * g[e] + bb[e];
    vout[row * EDIM + e] = (bf16)y;
    float pe = 0.f;
    if (valid) {
      int sec = e >> 9, j = e & 511;
      float omega = __expf(-0.0179889460390160f * (float)j);  // ln(1e4)/512
      float ang = (float)((sec < 2) ? rpos : cpos) * omega;
      pe = (sec & 1) ? __cosf(ang) : __sinf(ang);
    }
    kout[row * EDIM + e] = (bf16)(y + pe);
  }
}

// ------------------------------ softmax ------------------------------------
__global__ void k_softmax(const float* __restrict__ sc, const int* __restrict__ tgt,
                          bf16* __restrict__ attn) {
  __shared__ float sbuf[256];
  long row = blockIdx.x;                // (b*16+h)*64+q ; b = row>>10
  int b = (int)(row >> 10);
  int tw = tgt[2 * b + 1]; tw = tw < 1 ? 1 : (tw > 32 ? 32 : tw);
  int vl = tgt[2 * b] * tw;
  const float* x = sc + row * 1024;
  float v[4];
#pragma unroll
  for (int t = 0; t < 4; t++) {
    int i = threadIdx.x + t * 256;
    v[t] = (i < vl) ? x[i] : -1e9f;
  }
  float m = blk_max(fmaxf(fmaxf(v[0], v[1]), fmaxf(v[2], v[3])), sbuf);
  float e[4];
  float s = 0.f;
#pragma unroll
  for (int t = 0; t < 4; t++) { e[t] = __expf(v[t] - m); s += e[t]; }
  float inv = 1.0f / blk_sum(s, sbuf);
#pragma unroll
  for (int t = 0; t < 4; t++)
    attn[row * 1024 + threadIdx.x + t * 256] = (bf16)(e[t] * inv);
}

// --------------------------- WMMA GEMM common ------------------------------
// C[M,N] = scale * (A[M,K] . B[N,K]^T) + bias[n]
struct GemmP {
  const bf16* A; const bf16* B; void* C; const float* bias;
  long lda, ldb, ldc;
  int  K;
  float scale;
  long A_db, A_dh, B_db, B_dh, C_db, C_dh;   // z = b*16+h offsets (elements)
};

__device__ __forceinline__ v16bf load_a16(const bf16* pa) {
  v16bf f;
  *(f32x4*)&f       = *(const f32x4*)(pa);
  *((f32x4*)&f + 1) = *(const f32x4*)(pa + 8);
  return f;
}

// epilogue: C/D layout — VGPR r: lanes 0-15 -> M=r, lanes 16-31 -> M=r+8
template<int EPI>
__device__ __forceinline__ void store_tile(const GemmP& p, f32x8 (&acc)[2][4],
                                           int m0, int n0, long cbase, int lane) {
  const int mh = (lane >> 4) << 3;
#pragma unroll
  for (int i = 0; i < 2; i++) {
#pragma unroll
    for (int j = 0; j < 4; j++) {
      const int n = n0 + j * 16 + (lane & 15);
      const float bv = p.bias ? p.bias[n] : 0.0f;
#pragma unroll
      for (int r = 0; r < 8; r++) {
        const int m = m0 + i * 16 + r + mh;
        const float val = acc[i][j][r] * p.scale + bv;
        if (EPI == 0) {
          ((bf16*)p.C)[cbase + (long)m * p.ldc + n] = (bf16)val;
        } else if (EPI == 1) {
          ((float*)p.C)[cbase + (long)m * p.ldc + n] = val;
        } else {  // vhT transposed store: m = b*1024+l
          const long idx = ((long)(m >> 10) << 21) + ((long)n << 10) + (long)(m & 1023);
          ((bf16*)p.C)[idx] = (bf16)val;
        }
      }
    }
  }
}

// ----------------- direct-global GEMM (attention, batched) -----------------
template<int WM, int WN, int EPI>
__global__ void k_gemm(GemmP p) {
  const int lane = threadIdx.x & 31;
  const int wave = threadIdx.x >> 5;
  const int wm = wave % WM, wn = wave / WM;
  const long zb = blockIdx.z >> 4, zh = blockIdx.z & 15;

  const bf16* A = p.A + zb * p.A_db + zh * p.A_dh;
  const bf16* B = p.B + zb * p.B_db + zh * p.B_dh;
  const long cbase = zb * p.C_db + zh * p.C_dh;

  const int m0 = blockIdx.y * (WM * 32) + wm * 32;
  const int n0 = blockIdx.x * (WN * 64) + wn * 64;
  const bf16* Ab = A + (long)m0 * p.lda;
  const bf16* Bb = B + (long)n0 * p.ldb;

  f32x8 acc[2][4];
#pragma unroll
  for (int i = 0; i < 2; i++)
#pragma unroll
    for (int j = 0; j < 4; j++) acc[i][j] = (f32x8)0.0f;

  const int ar  = lane & 15;
  const int akg = (lane >> 4) << 4;    // A: K sub-offset 0 or 16
  const int bkg = (lane >> 4) << 4;    // B: K sub-offset 0 or 16

  for (int k = 0; k < p.K; k += 32) {
    v16bf af[2], bfr[4];
#pragma unroll
    for (int i = 0; i < 2; i++)
      af[i] = load_a16(Ab + (long)(ar + i * 16) * p.lda + k + akg);
#pragma unroll
    for (int j = 0; j < 4; j++) {
      const bf16* pb = Bb + (long)(ar + j * 16) * p.ldb + k + bkg;
      v16bf f;
      *(f32x4*)&f       = *(const f32x4*)(pb);
      *((f32x4*)&f + 1) = *(const f32x4*)(pb + 8);
      bfr[j] = f;
    }
    if (k + 32 < p.K) {
      __builtin_prefetch(Ab + (long)ar * p.lda + k + 32 + akg, 0, 1);
      __builtin_prefetch(Bb + (long)ar * p.ldb + k + 32 + bkg, 0, 1);
    }
#pragma unroll
    for (int i = 0; i < 2; i++)
#pragma unroll
      for (int j = 0; j < 4; j++)
        acc[i][j] = __builtin_amdgcn_wmma_f32_16x16x32_bf16(
            false, af[i], false, bfr[j], (short)0, acc[i][j], false, false);
  }
  store_tile<EPI>(p, acc, m0, n0, cbase, lane);
}

// --------- TDM-staged GEMM: B tile -> LDS via tensor_load_to_lds -----------
// LDS image per k-step: BN rows x 32 bf16, TDM pad inserts 16B every 64B
// -> 80B row pitch (20-bank stride, 16B aligned, conflict-free ds_load_b128).
template<int WM, int WN, int EPI>
__global__ void k_gemm_tdm(GemmP p) {
  constexpr int BN   = WN * 64;          // B rows staged per block
  constexpr int ROWB = 80;               // padded LDS row pitch (bytes)
  constexpr int BUFB = BN * ROWB;        // one buffer
  __shared__ char smemB[2 * BUFB];

  const int lane = threadIdx.x & 31;
  const int wave = threadIdx.x >> 5;
  const int wm = wave % WM, wn = wave / WM;

  const int m0    = blockIdx.y * (WM * 32) + wm * 32;
  const int n0blk = blockIdx.x * BN;
  const bf16* Ab   = p.A + (long)m0 * p.lda;
  const bf16* Bblk = p.B + (long)n0blk * p.ldb;

  const int KT = p.K >> 5;
  const unsigned ldsBase = (unsigned)(size_t)(&smemB[0]);   // low 32b = LDS offset

  // D# group1 constants (ISA 8.4): mask=0, data_size=1(2B), pad_enable,
  // pad_interval=3 (16 dwords = 64B), pad_amount=3 (4 dwords = 16B)
  const unsigned g1w0 = (1u << 16) | (1u << 20) | (3u << 22) | (3u << 25);
  const unsigned td0 = 1u << 20, td1 = 1u << 20;   // tensor dims (no OOB by construction)
  const unsigned long long st0 = (unsigned long long)p.ldb;  // dim0 stride (elements)
  const unsigned g1w1 = (td0 & 0xFFFFu) << 16;
  const unsigned g1w2 = (td0 >> 16) | ((td1 & 0xFFFFu) << 16);
  const unsigned g1w3 = (td1 >> 16) | (32u << 16);           // tile_dim0 = 32
  const unsigned g1w4 = (unsigned)BN;                        // tile_dim1 = BN, tile_dim2 = 0
  const unsigned g1w5 = (unsigned)(st0 & 0xFFFFFFFFull);
  const unsigned g1w6 = (unsigned)(st0 >> 32);               // + stride1[15:0]=0
  const i32x4 gz4 = {0, 0, 0, 0};
  const i32x8 gz8 = {0, 0, 0, 0, 0, 0, 0, 0};

  auto tdm_issue = [&](int kt, int bufsel) {
    unsigned long long ga = (unsigned long long)(size_t)Bblk + (unsigned long long)kt * 64ull;
    u32x4 g0 = { 1u,                                         // count=1, is_restore=0
                 ldsBase + (unsigned)(bufsel * BUFB),        // lds_addr
                 (unsigned)ga,                               // global_addr[31:0]
                 ((unsigned)(ga >> 32) & 0x1FFFFFFu) | (2u << 30) };  // [56:32] | type=2
    i32x8 g1 = { (int)g1w0, (int)g1w1, (int)g1w2, (int)g1w3,
                 (int)g1w4, (int)g1w5, (int)g1w6, 0 };
    __builtin_amdgcn_tensor_load_to_lds(g0, g1, gz4, gz4, gz8, 0);
  };

  if (wave == 0) tdm_issue(0, 0);

  f32x8 acc[2][4];
#pragma unroll
  for (int i = 0; i < 2; i++)
#pragma unroll
    for (int j = 0; j < 4; j++) acc[i][j] = (f32x8)0.0f;

  const int ar  = lane & 15;
  const int akg = (lane >> 4) << 4;
  const int bkg = (lane >> 4) << 4;

  for (int kt = 0; kt < KT; ++kt) {
    if (wave == 0) {
      if (kt + 1 < KT) {
        tdm_issue(kt + 1, (kt + 1) & 1);
        __builtin_amdgcn_s_wait_tensorcnt((short)1);   // current buffer landed
      } else {
        __builtin_amdgcn_s_wait_tensorcnt((short)0);
      }
    }
    __syncthreads();                                   // buffer visible to all waves

    const char* bbuf = &smemB[(kt & 1) * BUFB];
    const int k = kt * 32;
    v16bf af[2], bfr[4];
#pragma unroll
    for (int i = 0; i < 2; i++)
      af[i] = load_a16(Ab + (long)(ar + i * 16) * p.lda + k + akg);
#pragma unroll
    for (int j = 0; j < 4; j++) {
      const int n_l = wn * 64 + j * 16 + ar;
      const f32x4* qp = (const f32x4*)(bbuf + n_l * ROWB + bkg * 2);
      v16bf f;
      *(f32x4*)&f       = qp[0];
      *((f32x4*)&f + 1) = qp[1];
      bfr[j] = f;
    }
    if (kt + 1 < KT)
      __builtin_prefetch(Ab + (long)ar * p.lda + k + 32 + akg, 0, 1);
#pragma unroll
    for (int i = 0; i < 2; i++)
#pragma unroll
      for (int j = 0; j < 4; j++)
        acc[i][j] = __builtin_amdgcn_wmma_f32_16x16x32_bf16(
            false, af[i], false, bfr[j], (short)0, acc[i][j], false, false);
    __syncthreads();                                   // all reads done before refill
  }
  store_tile<EPI>(p, acc, m0, n0blk + wn * 64, 0, lane);
}

// ------------------------------- launcher ----------------------------------
extern "C" void kernel_launch(void* const* d_in, const int* in_sizes, int n_in,
                              void* d_out, int out_size, void* d_ws, size_t ws_size,
                              hipStream_t stream) {
  const float* img   = (const float*)d_in[0];
  const int*   tgt   = (const int*)  d_in[1];
  const float* query = (const float*)d_in[2];
  const float* kv_w  = (const float*)d_in[3];
  const float* lnq_g = (const float*)d_in[4];
  const float* lnq_b = (const float*)d_in[5];
  const float* lnkv_g= (const float*)d_in[6];
  const float* lnkv_b= (const float*)d_in[7];
  const float* lnp_g = (const float*)d_in[8];
  const float* lnp_b = (const float*)d_in[9];
  const float* wq    = (const float*)d_in[10];
  const float* bq    = (const float*)d_in[11];
  const float* wk    = (const float*)d_in[12];
  const float* bk    = (const float*)d_in[13];
  const float* wv    = (const float*)d_in[14];
  const float* bv    = (const float*)d_in[15];
  const float* wo    = (const float*)d_in[16];
  const float* bo    = (const float*)d_in[17];
  const float* proj  = (const float*)d_in[18];

  char* ws = (char*)d_ws;
  bf16* imgBF = (bf16*)(ws + OFF_IMGBF);
  bf16* kvwBF = (bf16*)(ws + OFF_KVWBF);
  bf16* wqBF  = (bf16*)(ws + OFF_WQ);
  bf16* wkBF  = (bf16*)(ws + OFF_WK);
  bf16* wvBF  = (bf16*)(ws + OFF_WV);
  bf16* woBF  = (bf16*)(ws + OFF_WO);
  bf16* pjtBF = (bf16*)(ws + OFF_PJT);
  bf16* qlnBF = (bf16*)(ws + OFF_QLN);
  bf16* qhBF  = (bf16*)(ws + OFF_QH);
  bf16* xBF   = (bf16*)(ws + OFF_S1);
  bf16* khBF  = (bf16*)(ws + OFF_S1);     // reuses xBF after ln_kv
  bf16* kBF   = (bf16*)(ws + OFF_S2);
  float* scoresF = (float*)(ws + OFF_S2); // reuses kBF after GEMM2k
  bf16* vBF   = (bf16*)(ws + OFF_S3);
  bf16* attnBF= (bf16*)(ws + OFF_S3);     // reuses vBF after GEMM2v
  bf16* vhT   = (bf16*)(ws + OFF_S4);
  bf16* oBF   = (bf16*)(ws + OFF_OBF);    // reuses imgBF region
  float* o2F  = (float*)(ws + OFF_O2F);
  bf16* o2ln  = (bf16*)(ws + OFF_O2LN);

  // 1. conversions
  k_cvt_bf16<<<8192, 256, 0, stream>>>(img,  imgBF, (long)BATCH * LSEQ * VPM);
  k_cvt_bf16<<<2048, 256, 0, stream>>>(kv_w, kvwBF, (long)EDIM * VPM);
  k_cvt_bf16<<<4096, 256, 0, stream>>>(wq, wqBF, (long)EDIM * EDIM);
  k_cvt_bf16<<<4096, 256, 0, stream>>>(wk, wkBF, (long)EDIM * EDIM);
  k_cvt_bf16<<<4096, 256, 0, stream>>>(wv, wvBF, (long)EDIM * EDIM);
  k_cvt_bf16<<<4096, 256, 0, stream>>>(wo, woBF, (long)EDIM * EDIM);
  k_cvt_transpose<<<dim3(64, 64), 256, 0, stream>>>(proj, pjtBF);

  GemmP p;
  p.A_db = p.A_dh = p.B_db = p.B_dh = p.C_db = p.C_dh = 0;
  p.scale = 1.0f;

  // 2. q path: ln(query) -> qln ; qh = qln @ wq^T + bq
  k_ln_f32<<<NQ, 256, 0, stream>>>(query, lnq_g, lnq_b, qlnBF);
  p.A = qlnBF; p.lda = EDIM; p.B = wqBF; p.ldb = EDIM;
  p.C = qhBF;  p.ldc = EDIM; p.bias = bq; p.K = EDIM;
  k_gemm_tdm<2, 4, 0><<<dim3(8, 1, 1), 256, 0, stream>>>(p);

  // 3. x = img @ kv_w^T
  p.A = imgBF; p.lda = VPM; p.B = kvwBF; p.ldb = VPM;
  p.C = xBF;   p.ldc = EDIM; p.bias = nullptr; p.K = VPM;
  k_gemm_tdm<4, 2, 0><<<dim3(16, 256, 1), 256, 0, stream>>>(p);

  // 4. ln_kv + pos-emb: v = ln(x), k = ln(x)+pe
  k_ln_kv_pe<<<BATCH * LSEQ, 256, 0, stream>>>(xBF, lnkv_g, lnkv_b, tgt, vBF, kBF);

  // 5a. kh = k @ wk^T + bk
  p.A = kBF; p.lda = EDIM; p.B = wkBF; p.ldb = EDIM;
  p.C = khBF; p.ldc = EDIM; p.bias = bk; p.K = EDIM;
  k_gemm_tdm<4, 2, 0><<<dim3(16, 256, 1), 256, 0, stream>>>(p);

  // 5b. vhT[b, n, l] = (v @ wv^T + bv)[b*1024+l, n]   (transposed store)
  p.A = vBF; p.B = wvBF; p.C = vhT; p.bias = bv;
  k_gemm_tdm<4, 2, 2><<<dim3(16, 256, 1), 256, 0, stream>>>(p);

  // 6. scores[b,h,q,l] = qh . kh / sqrt(128)
  p.A = qhBF;  p.lda = EDIM; p.A_db = 0;                     p.A_dh = 128;
  p.B = khBF;  p.ldb = EDIM; p.B_db = (long)LSEQ * EDIM;     p.B_dh = 128;
  p.C = scoresF; p.ldc = 1024; p.C_db = 16L * 64 * 1024;     p.C_dh = 64L * 1024;
  p.bias = nullptr; p.K = 128; p.scale = 0.08838834764831845f;
  k_gemm<2, 4, 1><<<dim3(4, 1, BATCH * NH), 256, 0, stream>>>(p);

  // 7. masked softmax -> attn bf16
  k_softmax<<<BATCH * NH * NQ, 256, 0, stream>>>(scoresF, tgt, attnBF);

  // 8. o[b,q,h*128+d] = attn[b,h] @ vh[b,h]
  p.A = attnBF; p.lda = 1024; p.A_db = 16L * 64 * 1024;      p.A_dh = 64L * 1024;
  p.B = vhT;    p.ldb = 1024; p.B_db = (long)EDIM * 1024;    p.B_dh = 128L * 1024;
  p.C = oBF;    p.ldc = EDIM; p.C_db = 64L * EDIM;           p.C_dh = 128;
  p.bias = nullptr; p.K = 1024; p.scale = 1.0f;
  k_gemm<2, 2, 0><<<dim3(1, 1, BATCH * NH), 128, 0, stream>>>(p);

  // 9a. o2 = o @ wo^T + bo  (f32)
  p.A_db = p.A_dh = p.B_db = p.B_dh = p.C_db = p.C_dh = 0;
  p.A = oBF; p.lda = EDIM; p.B = woBF; p.ldb = EDIM;
  p.C = o2F; p.ldc = EDIM; p.bias = bo; p.K = EDIM;
  k_gemm_tdm<4, 2, 1><<<dim3(16, 16, 1), 256, 0, stream>>>(p);

  // 9b. ln_post
  k_ln_f32<<<BATCH * NQ, 256, 0, stream>>>(o2F, lnp_g, lnp_b, o2ln);

  // 9c. out = o2ln @ proj  (= o2ln @ projT^T), f32 -> d_out
  p.A = o2ln; p.lda = EDIM; p.B = pjtBF; p.ldb = EDIM;
  p.C = d_out; p.ldc = EDIM; p.bias = nullptr; p.K = EDIM;
  k_gemm_tdm<4, 2, 1><<<dim3(16, 16, 1), 256, 0, stream>>>(p);
}
